// MultiHeadSelfAttention_75265006895818
// MI455X (gfx1250) — matrix operand order repaired and checked by reading, not compiled, not run on hardware
//
#include <hip/hip_runtime.h>
#include <hip/hip_bf16.h>

typedef _Float16 half_t;
typedef __attribute__((ext_vector_type(16))) _Float16 v16h;
typedef __attribute__((ext_vector_type(8)))  _Float16 v8h;
typedef __attribute__((ext_vector_type(8)))  float    v8f;
typedef unsigned int u32x4 __attribute__((ext_vector_type(4)));
typedef int          i32x4 __attribute__((ext_vector_type(4)));
typedef int          i32x8 __attribute__((ext_vector_type(8)));

#define D_MODEL   1024
#define NUM_HEADS 16
#define HEAD_DIM  64
#define SEQ       2048
#define BATCH     4
#define M_TOTAL   (BATCH * SEQ)

// ---------------------------------------------------------------------------
// WMMA helper: D = A(16x32 f16) * B(32x16 f16) + C(16x16 f32)
// ---------------------------------------------------------------------------
__device__ __forceinline__ v8f wmma_f16(v16h a, v16h b, v8f c) {
  return __builtin_amdgcn_wmma_f32_16x16x32_f16(false, a, false, b, (short)0, c,
                                                false, false);
}

// ---------------------------------------------------------------------------
// CDNA5 async copy: global -> LDS, 16 B per lane per issue (ASYNCcnt tracked)
// ---------------------------------------------------------------------------
__device__ __forceinline__ void async_b128(unsigned lds_off, const void* gaddr) {
  asm volatile("global_load_async_to_lds_b128 %0, %1, off"
               :: "v"(lds_off), "v"(gaddr) : "memory");
}
__device__ __forceinline__ void wait_async0() {
  asm volatile("s_wait_asynccnt 0" ::: "memory");
}
__device__ __forceinline__ void wait_ds0() {
  asm volatile("s_wait_dscnt 0" ::: "memory");
}

// ---------------------------------------------------------------------------
// CDNA5 Tensor Data Mover: one descriptor moves a 32x64 f16 tile to LDS.
// D# per CDNA5 ISA (2D, data_size=2B, dims 64x2048,
// tile 64x32, stride 64; groups 2/3 zero).  Tracked by TENSORcnt.
// ---------------------------------------------------------------------------
__device__ __forceinline__ void tdm_load_2d(const half_t* gsrc, half_t* ldst) {
  const unsigned lds  = __builtin_amdgcn_readfirstlane((unsigned)(uintptr_t)ldst);
  const unsigned long long ga = (unsigned long long)(uintptr_t)gsrc;
  const unsigned galo = __builtin_amdgcn_readfirstlane((unsigned)ga);
  const unsigned gahi = __builtin_amdgcn_readfirstlane((unsigned)(ga >> 32));
  const u32x4 g0 = { 1u,                                   // count = 1 (valid)
                     lds,                                  // lds_addr
                     galo,                                 // global_addr[31:0]
                     (gahi & 0x01FFFFFFu) | (2u << 30) };  // ga[56:32] | type=2
  const i32x8 g1 = { (1 << 16),                 // data_size = 2 B
                     (HEAD_DIM & 0xFFFF) << 16, // tensor_dim0 = 64
                     SEQ << 16,                 // tensor_dim1 = 2048
                     HEAD_DIM << 16,            // tile_dim0 = 64
                     32,                        // tile_dim1 = 32 rows
                     HEAD_DIM,                  // tensor_dim0_stride = 64
                     0, 0 };
  const i32x4 z4 = { 0, 0, 0, 0 };
#if __clang_major__ >= 23
  const i32x8 z8 = { 0, 0, 0, 0, 0, 0, 0, 0 };
  __builtin_amdgcn_tensor_load_to_lds(g0, g1, z4, z4, z8, 0);
#else
  __builtin_amdgcn_tensor_load_to_lds(g0, g1, z4, z4, 0);
#endif
}

// ---------------------------------------------------------------------------
// Fragment loaders (wave32; A and B share a contiguous k order chosen here)
// A (16x32): lane&15 -> row, K = 16 contiguous at kb = 16*(lane/16)
// B (32x16): lane&15 -> col, K = 16 contiguous at kb = 16*(lane/16)
// C (16x16): lane&15 -> col, VGPR r -> row r + 8*(lane/16)
// ---------------------------------------------------------------------------
__device__ __forceinline__ v16h a_frag_f16(const half_t* line) {
  const int kb = ((threadIdx.x >> 4) & 1) * 16;
  const v8h lo = *(const v8h*)(line + kb);
  const v8h hi = *(const v8h*)(line + kb + 8);
  v16h f;
#pragma unroll
  for (int i = 0; i < 8; ++i) { f[i] = lo[i]; f[i + 8] = hi[i]; }
  return f;
}

__device__ __forceinline__ v16h b_frag_f16(const half_t* line) {
  const int kb = ((threadIdx.x >> 4) & 1) * 16;
  const v8h lo = *(const v8h*)(line + kb);
  const v8h hi = *(const v8h*)(line + kb + 8);
  v16h f;
#pragma unroll
  for (int i = 0; i < 8; ++i) { f[i] = lo[i]; f[i + 8] = hi[i]; }
  return f;
}

// V as B-fragment out of the LDS-staged 32x64 block: element(k,n) = vbuf[k][nf]
__device__ __forceinline__ v16h v_frag_lds(const half_t* vbuf, int nf) {
  const int kb = ((threadIdx.x >> 4) & 1) * 16;
  v16h f;
#pragma unroll
  for (int i = 0; i < 16; ++i)
    f[i] = vbuf[(kb + i) * HEAD_DIM + nf];
  return f;
}

// ---------------------------------------------------------------------------
// Kernel 0: f32 -> f16 conversion (8 elems/thread), optional scale fold
// ---------------------------------------------------------------------------
__global__ void __launch_bounds__(256)
cvt_f16(const float* __restrict__ src, half_t* __restrict__ dst, float scale,
        int nelem8) {
  const int i = blockIdx.x * 256 + threadIdx.x;
  if (i >= nelem8) return;
  const float4* s = (const float4*)src + (size_t)i * 2;
  const float4 a = s[0], b = s[1];
  v8h o;
  o[0] = (half_t)(a.x * scale); o[1] = (half_t)(a.y * scale);
  o[2] = (half_t)(a.z * scale); o[3] = (half_t)(a.w * scale);
  o[4] = (half_t)(b.x * scale); o[5] = (half_t)(b.y * scale);
  o[6] = (half_t)(b.z * scale); o[7] = (half_t)(b.w * scale);
  *(v8h*)(dst + (size_t)i * 8) = o;
}

// ---------------------------------------------------------------------------
// Kernel 1: one of Q/K/V projection with fused RoPE.  One wave computes a
// 32x64 output tile (one head wide): 2 A-frags x 4 B-frags -> 8 WMMAs per
// k-step, every fragment reused.  out: [B, H, S, HEAD_DIM] f16.
// mode: 0/1 = Q/K (rope), 2 = V.  (1/sqrt(hd) folded into Wq conversion.)
// ---------------------------------------------------------------------------
__global__ void __launch_bounds__(32)
qkv_rope_gemm(const half_t* __restrict__ XH, const half_t* __restrict__ WH,
              const int* __restrict__ pos, half_t* __restrict__ out, int mode) {
  const int mbase = blockIdx.x * 32;
  const int h     = blockIdx.y;            // 64-col tile == one head
  const int lane  = threadIdx.x & 31;
  const int n     = lane & 15;

  const half_t* arow0 = XH + (size_t)(mbase + n) * D_MODEL;
  const half_t* arow1 = arow0 + (size_t)16 * D_MODEL;
  const half_t* b0 = WH + (size_t)(h * HEAD_DIM +  0 + n) * D_MODEL;
  const half_t* b1 = WH + (size_t)(h * HEAD_DIM + 16 + n) * D_MODEL;
  const half_t* b2 = WH + (size_t)(h * HEAD_DIM + 32 + n) * D_MODEL;
  const half_t* b3 = WH + (size_t)(h * HEAD_DIM + 48 + n) * D_MODEL;

  v8f c0[4] = {}, c1[4] = {};
#pragma unroll 2
  for (int k = 0; k < D_MODEL; k += 32) {
    __builtin_prefetch(arow0 + k + 64, 0, 3);   // global_prefetch_b8
    __builtin_prefetch(b0 + k + 64, 0, 3);
    const v16h a0 = a_frag_f16(arow0 + k);
    const v16h a1 = a_frag_f16(arow1 + k);
    v16h b;
    b = b_frag_f16(b0 + k); c0[0] = wmma_f16(a0, b, c0[0]); c1[0] = wmma_f16(a1, b, c1[0]);
    b = b_frag_f16(b1 + k); c0[1] = wmma_f16(a0, b, c0[1]); c1[1] = wmma_f16(a1, b, c1[1]);
    b = b_frag_f16(b2 + k); c0[2] = wmma_f16(a0, b, c0[2]); c1[2] = wmma_f16(a1, b, c1[2]);
    b = b_frag_f16(b3 + k); c0[3] = wmma_f16(a0, b, c0[3]); c1[3] = wmma_f16(a1, b, c1[3]);
  }

  const int bidx  = mbase / SEQ;             // tiles never straddle a batch
  const int sbase = mbase % SEQ;
  half_t* obase = out + ((size_t)bidx * NUM_HEADS + h) * (size_t)SEQ * HEAD_DIM;

#pragma unroll
  for (int mt = 0; mt < 2; ++mt) {
    const v8f* cc = mt ? c1 : c0;
    const int sb = sbase + mt * 16;
#pragma unroll
    for (int ct = 0; ct < 4; ++ct) {
      const int hd = ct * 16 + n;            // feature index within head
      float inv_freq = 0.0f;
      if (mode < 2)  // 10000^(-2i/64) = exp(-i * ln(10000)/32)
        inv_freq = __expf(-(float)(hd >> 1) * 0.28782313662425572f);
#pragma unroll
      for (int r = 0; r < 8; ++r) {
        const int s = sb + r + 8 * (lane >> 4);
        float v = cc[ct][r];
        if (mode < 2) {
          const float other = __shfl_xor(v, 1, 32);  // RoPE partner in lane^1
          float sn, cs;
          __sincosf((float)pos[s] * inv_freq, &sn, &cs);
          v = v * cs + ((hd & 1) ? other : -other) * sn;
        }
        obase[(size_t)s * HEAD_DIM + hd] = (half_t)v;
      }
    }
  }
}

// ---------------------------------------------------------------------------
// Kernel 2: causal flash attention.  One wave per (b, h, 16-row q tile).
// K and V blocks are double-buffered in LDS: K via async-to-LDS b128,
// V via one TDM tensor_load_to_lds descriptor per block; next block's
// staging is issued before computing the current block so the DMA overlaps
// the QK^T WMMAs + softmax.
// ---------------------------------------------------------------------------
__global__ void __launch_bounds__(128)
flash_attn(const half_t* __restrict__ Q, const half_t* __restrict__ K,
           const half_t* __restrict__ V, half_t* __restrict__ O) {
  const int wid   = threadIdx.x >> 5;
  const int lane  = threadIdx.x & 31;
  const int qt    = blockIdx.x * 4 + wid;
  const int h     = blockIdx.y;
  const int b     = blockIdx.z;
  const int qbase = qt * 16;
  const int n     = lane & 15;
  const int hsel  = lane >> 4;

  const size_t bh = ((size_t)b * NUM_HEADS + h) * (size_t)SEQ * HEAD_DIM;
  const half_t* Qb = Q + bh;
  const half_t* Kb = K + bh;
  const half_t* Vb = V + bh;

  __shared__ alignas(16) half_t pst[4][16 * 32];               // 1 KB/wave
  __shared__ alignas(16) half_t kst[4][2][32 * HEAD_DIM];      // 8 KB/wave
  __shared__ alignas(16) half_t vst[4][2][32 * HEAD_DIM];      // 8 KB/wave
  half_t* pbuf = pst[wid];

  const half_t* qrow = Qb + (size_t)(qbase + n) * HEAD_DIM;
  const v16h qa0 = a_frag_f16(qrow);        // head-dim 0..31
  const v16h qa1 = a_frag_f16(qrow + 32);   // head-dim 32..63

  v8f acc0 = {}, acc1 = {}, acc2 = {}, acc3 = {};
  float mrow[8], lrow[8];
#pragma unroll
  for (int r = 0; r < 8; ++r) { mrow[r] = -1e30f; lrow[r] = 0.0f; }

  // stage one 32-token block: K async, V via TDM (blocks are contiguous 4 KB;
  // last block always ends at row <= SEQ-1, so no clamping is needed)
  auto stage = [&](int kb0, int sel) {
    wait_ds0();   // prior reads of this buffer pair must have landed
    const half_t* kg = Kb + (size_t)kb0 * HEAD_DIM;
    half_t* kl = kst[wid][sel];
#pragma unroll
    for (int i = 0; i < 8; ++i) {
      const int chunk = i * 32 + lane;      // 256 x 16 B chunks
      async_b128((unsigned)(uintptr_t)(kl + chunk * 8), kg + chunk * 8);
    }
    tdm_load_2d(Vb + (size_t)kb0 * HEAD_DIM, vst[wid][sel]);
  };

  const int kend = qbase + 16;              // causal horizon
  stage(0, 0);
  int sel = 0;
  for (int kb0 = 0; kb0 < kend; kb0 += 32, sel ^= 1) {
    wait_async0();                          // K block resident
    __builtin_amdgcn_s_wait_tensorcnt(0);   // V block resident
    if (kb0 + 32 < kend) stage(kb0 + 32, sel ^ 1);

    const half_t* kbuf = kst[wid][sel];
    const half_t* vbuf = vst[wid][sel];

    // ---- scores: two 16x16 tiles over this 32-token block (K from LDS) ----
    v8f s0 = {}, s1 = {};
    const half_t* kl0 = kbuf + n * HEAD_DIM;
    s0 = wmma_f16(qa0, b_frag_f16(kl0), s0);
    s0 = wmma_f16(qa1, b_frag_f16(kl0 + 32), s0);
    const half_t* kl1 = kbuf + (16 + n) * HEAD_DIM;
    s1 = wmma_f16(qa0, b_frag_f16(kl1), s1);
    s1 = wmma_f16(qa1, b_frag_f16(kl1 + 32), s1);

    // ---- causal mask + online softmax (rows share the C-layout mapping) ----
#pragma unroll
    for (int r = 0; r < 8; ++r) {
      const int mm   = r + 8 * hsel;
      const int qtok = qbase + mm;
      float v0 = (kb0 + n      <= qtok) ? s0[r] : -1e30f;
      float v1 = (kb0 + 16 + n <= qtok) ? s1[r] : -1e30f;
      float bm = fmaxf(v0, v1);
#pragma unroll
      for (int off = 1; off < 16; off <<= 1) bm = fmaxf(bm, __shfl_xor(bm, off, 32));
      const float nm   = fmaxf(mrow[r], bm);
      const float corr = __expf(mrow[r] - nm);
      mrow[r] = nm;
      const float p0 = __expf(v0 - nm);
      const float p1 = __expf(v1 - nm);
      float ps = p0 + p1;
#pragma unroll
      for (int off = 1; off < 16; off <<= 1) ps += __shfl_xor(ps, off, 32);
      lrow[r] = lrow[r] * corr + ps;
      acc0[r] *= corr; acc1[r] *= corr; acc2[r] *= corr; acc3[r] *= corr;
      pbuf[mm * 32 + n]      = (half_t)p0;   // stage P (row-major 16x32 f16)
      pbuf[mm * 32 + 16 + n] = (half_t)p1;
    }

    // ---- O += P V  (P from LDS as A-frag; V from TDM-staged LDS) ----
    const v16h pa = a_frag_f16(pbuf + n * 32);
    acc0 = wmma_f16(pa, v_frag_lds(vbuf, n),      acc0);
    acc1 = wmma_f16(pa, v_frag_lds(vbuf, 16 + n), acc1);
    acc2 = wmma_f16(pa, v_frag_lds(vbuf, 32 + n), acc2);
    acc3 = wmma_f16(pa, v_frag_lds(vbuf, 48 + n), acc3);
  }

  // ---- normalize and store O tile as f16 into [B, S, D_MODEL] ----
#pragma unroll
  for (int r = 0; r < 8; ++r) {
    const int mm = r + 8 * hsel;
    const float inv = 1.0f / lrow[r];
    half_t* orow = O + ((size_t)b * SEQ + qbase + mm) * D_MODEL + h * HEAD_DIM;
    orow[n]      = (half_t)(acc0[r] * inv);
    orow[16 + n] = (half_t)(acc1[r] * inv);
    orow[32 + n] = (half_t)(acc2[r] * inv);
    orow[48 + n] = (half_t)(acc3[r] * inv);
  }
}

// ---------------------------------------------------------------------------
// Kernel 3: output projection  out = O @ Wo^T  (f16 A and B, f32 out)
// One wave per 32x64 output tile (same blocking as the QKV GEMM).
// ---------------------------------------------------------------------------
__global__ void __launch_bounds__(32)
out_proj_gemm(const half_t* __restrict__ A, const half_t* __restrict__ WoH,
              float* __restrict__ out) {
  const int mbase = blockIdx.x * 32;
  const int nbase = blockIdx.y * 64;
  const int lane  = threadIdx.x & 31;
  const int n     = lane & 15;

  const half_t* arow0 = A + (size_t)(mbase + n) * D_MODEL;
  const half_t* arow1 = arow0 + (size_t)16 * D_MODEL;
  const half_t* b0 = WoH + (size_t)(nbase +  0 + n) * D_MODEL;
  const half_t* b1 = WoH + (size_t)(nbase + 16 + n) * D_MODEL;
  const half_t* b2 = WoH + (size_t)(nbase + 32 + n) * D_MODEL;
  const half_t* b3 = WoH + (size_t)(nbase + 48 + n) * D_MODEL;

  v8f c0[4] = {}, c1[4] = {};
#pragma unroll 2
  for (int k = 0; k < D_MODEL; k += 32) {
    __builtin_prefetch(arow0 + k + 64, 0, 3);
    __builtin_prefetch(b0 + k + 64, 0, 3);
    const v16h a0 = a_frag_f16(arow0 + k);
    const v16h a1 = a_frag_f16(arow1 + k);
    v16h b;
    b = b_frag_f16(b0 + k); c0[0] = wmma_f16(a0, b, c0[0]); c1[0] = wmma_f16(a1, b, c1[0]);
    b = b_frag_f16(b1 + k); c0[1] = wmma_f16(a0, b, c0[1]); c1[1] = wmma_f16(a1, b, c1[1]);
    b = b_frag_f16(b2 + k); c0[2] = wmma_f16(a0, b, c0[2]); c1[2] = wmma_f16(a1, b, c1[2]);
    b = b_frag_f16(b3 + k); c0[3] = wmma_f16(a0, b, c0[3]); c1[3] = wmma_f16(a1, b, c1[3]);
  }

#pragma unroll
  for (int mt = 0; mt < 2; ++mt) {
    const v8f* cc = mt ? c1 : c0;
#pragma unroll
    for (int ct = 0; ct < 4; ++ct) {
#pragma unroll
      for (int r = 0; r < 8; ++r) {
        const int mm = mt * 16 + r + 8 * (lane >> 4);
        out[(size_t)(mbase + mm) * D_MODEL + nbase + ct * 16 + n] = cc[ct][r];
      }
    }
  }
}

// ---------------------------------------------------------------------------
extern "C" void kernel_launch(void* const* d_in, const int* in_sizes, int n_in,
                              void* d_out, int out_size, void* d_ws, size_t ws_size,
                              hipStream_t stream) {
  (void)in_sizes; (void)n_in; (void)out_size; (void)ws_size;
  const float* x   = (const float*)d_in[0];
  const float* Wq  = (const float*)d_in[1];
  const float* Wk  = (const float*)d_in[2];
  const float* Wv  = (const float*)d_in[3];
  const float* Wo  = (const float*)d_in[4];
  const int*   pos = (const int*)d_in[5];

  const size_t E  = (size_t)BATCH * NUM_HEADS * SEQ * HEAD_DIM;  // 8 Mi elems
  const size_t WE = (size_t)D_MODEL * D_MODEL;                   // 1 Mi elems
  half_t* qf  = (half_t*)d_ws;   // [B,H,S,64] f16, roped + pre-scaled
  half_t* kf  = qf + E;          // [B,H,S,64] f16, roped
  half_t* vf  = kf + E;          // [B,H,S,64] f16
  half_t* of  = vf + E;          // [B,S,1024] f16 attention output
  half_t* xh  = of + E;          // [B*S,1024] f16 input
  half_t* wqh = xh + (size_t)M_TOTAL * D_MODEL;
  half_t* wkh = wqh + WE;
  half_t* wvh = wkh + WE;
  half_t* woh = wvh + WE;

  // one-shot f16 staging (scale 1/sqrt(HEAD_DIM) folded into Wq)
  const int x8 = (int)((size_t)M_TOTAL * D_MODEL / 8), w8 = (int)(WE / 8);
  cvt_f16<<<(x8 + 255) / 256, 256, 0, stream>>>(x,  xh,  1.0f,   x8);
  cvt_f16<<<(w8 + 255) / 256, 256, 0, stream>>>(Wq, wqh, 0.125f, w8);
  cvt_f16<<<(w8 + 255) / 256, 256, 0, stream>>>(Wk, wkh, 1.0f,   w8);
  cvt_f16<<<(w8 + 255) / 256, 256, 0, stream>>>(Wv, wvh, 1.0f,   w8);
  cvt_f16<<<(w8 + 255) / 256, 256, 0, stream>>>(Wo, woh, 1.0f,   w8);

  dim3 gproj(M_TOTAL / 32, NUM_HEADS);
  qkv_rope_gemm<<<gproj, 32, 0, stream>>>(xh, wqh, pos, qf, 0);
  qkv_rope_gemm<<<gproj, 32, 0, stream>>>(xh, wkh, pos, kf, 1);
  qkv_rope_gemm<<<gproj, 32, 0, stream>>>(xh, wvh, pos, vf, 2);
  flash_attn<<<dim3(SEQ / 64, NUM_HEADS, BATCH), 128, 0, stream>>>(qf, kf, vf, of);
  out_proj_gemm<<<dim3(M_TOTAL / 32, D_MODEL / 64), 32, 0, stream>>>(of, woh,
                                                                     (float*)d_out);
}
